// SparseLlamaAttention_71253507441020
// MI455X (gfx1250) — matrix operand order repaired and checked by reading, not compiled, not run on hardware
//
#include <hip/hip_runtime.h>
#include <hip/hip_bf16.h>

typedef __bf16 bf16_t;
typedef __attribute__((ext_vector_type(32))) __bf16 v32bf;
typedef __attribute__((ext_vector_type(16))) __bf16 v16bf;
typedef __attribute__((ext_vector_type(8)))  __bf16 v8bf;
typedef __attribute__((ext_vector_type(8)))  float  v8f;

#define NUM_HEADS    32
#define NUM_KV_HEADS 8
#define HEAD_DIM     128
#define SEQ          2048
#define HID          4096
#define KVD          1024

// ---------- helpers ----------

__device__ __forceinline__ bf16_t f2bf(float f) {
  unsigned u = __builtin_bit_cast(unsigned, f);
  unsigned r = u + 0x7fffu + ((u >> 16) & 1u);   // round-to-nearest-even
  unsigned short h = (unsigned short)(r >> 16);
  return __builtin_bit_cast(bf16_t, h);
}

__device__ __forceinline__ v16bf cat8(v8bf lo, v8bf hi) {
  v16bf r;
#pragma unroll
  for (int i = 0; i < 8; ++i) { r[i] = lo[i]; r[i + 8] = hi[i]; }
  return r;
}

// A-operand (16x32 bf16) per-lane load.  Lane l: M = l&15, half h = l>>4.
// elements e=0..7 -> K = h*8 + e ; e=8..15 -> K = 16 + h*8 + (e-8).
// ptr must already include row*lda + h*8.
__device__ __forceinline__ v16bf load_a(const bf16_t* p) {
  v8bf lo = *(const v8bf*)(p);
  v8bf hi = *(const v8bf*)(p + 16);
  return cat8(lo, hi);
}

// B-operand for K=32 ops (32x16). Lane l: N = l&15, h = l>>4;
// e=0..7 -> K = h*8 + e ; e=8..15 -> K = 16 + h*8 + (e-8).  ptr includes row*ldb + h*8.
__device__ __forceinline__ v16bf load_b(const bf16_t* p) {
  v8bf lo = *(const v8bf*)(p);
  v8bf hi = *(const v8bf*)(p + 16);
  return cat8(lo, hi);
}

// B-operand for K=64 swmmac (64x16, v32bf). Lane l: N = l&15, h = l>>4;
// e=0..15 -> K = h*16 + e ; e=16..31 -> K = 32 + h*16 + (e-16).
// ptr includes row*ldb + h*16.
__device__ __forceinline__ v32bf load_b64(const bf16_t* p) {
  v8bf x0 = *(const v8bf*)(p);
  v8bf x1 = *(const v8bf*)(p + 8);
  v8bf x2 = *(const v8bf*)(p + 32);
  v8bf x3 = *(const v8bf*)(p + 40);
  v32bf r;
#pragma unroll
  for (int i = 0; i < 8; ++i) {
    r[i] = x0[i]; r[i + 8] = x1[i]; r[i + 16] = x2[i]; r[i + 24] = x3[i];
  }
  return r;
}

// ---------- kernel 0: recover permutation from one-hot matrix ----------
// perm = eye[p] => perm[c, p(c)] = 1; inv[j] = p^-1(j).
__global__ void invert_perm(const float* __restrict__ P, int* __restrict__ inv, int n) {
  int c = blockIdx.x * blockDim.x + threadIdx.x;
  if (c >= n) return;
  const float* row = P + (size_t)c * n;
  for (int j = 0; j < n; ++j)
    if (row[j] > 0.5f) { inv[j] = c; break; }
}

// ---------- kernel 1: 2:4 wanda prune -> compressed values + 2-bit indices ----
// One thread per (row, index-word of 8 groups).  Output in PERMUTED k-space:
//   Wc[o, 2g+s]  (s-th kept element of permuted group g, ascending position)
//   Idx[o, w]    8 nibbles, nibble gg = idx0 | idx1<<2  (idx0 < idx1)
__global__ void compress_weights(const float* __restrict__ W,
                                 const float* __restrict__ scaler,
                                 const int*   __restrict__ invp,
                                 bf16_t*      __restrict__ Wc,
                                 unsigned*    __restrict__ Idx,
                                 int O, int K) {
  int words = K >> 5;
  int t = blockIdx.x * blockDim.x + threadIdx.x;
  if (t >= O * words) return;
  int r = t / words, w = t % words;
  unsigned word = 0;
#pragma unroll
  for (int gg = 0; gg < 8; ++gg) {
    int g = w * 8 + gg;
    int   c[4]; float v[4], m[4];
#pragma unroll
    for (int b = 0; b < 4; ++b) {
      c[b] = invp[4 * g + b];
      v[b] = W[(size_t)r * K + c[b]];
      m[b] = fabsf(v[b]) * sqrtf(scaler[c[b]]);
    }
    unsigned i0 = 0, i1 = 0; float kv0 = 0.f, kv1 = 0.f; int slot = 0;
#pragma unroll
    for (int b = 0; b < 4; ++b) {
      int rank = 0;
#pragma unroll
      for (int b2 = 0; b2 < 4; ++b2)
        rank += (m[b2] > m[b]) || (m[b2] == m[b] && b2 < b);
      if (rank < 2) {
        if (slot == 0) { i0 = (unsigned)b; kv0 = v[b]; }
        else           { i1 = (unsigned)b; kv1 = v[b]; }
        ++slot;
      }
    }
    Wc[(size_t)r * (K >> 1) + 2 * g]     = f2bf(kv0);
    Wc[(size_t)r * (K >> 1) + 2 * g + 1] = f2bf(kv1);
    word |= (i0 | (i1 << 2)) << (4 * gg);
  }
  Idx[(size_t)r * words + w] = word;
}

// ---------- kernel 2: permute columns (+ optional f32->bf16) ----------
__global__ void permute_cvt_bf16(const float* __restrict__ X, const int* __restrict__ invp,
                                 bf16_t* __restrict__ Y, int rows, int cols) {
  int t = blockIdx.x * blockDim.x + threadIdx.x;
  if (t >= rows * cols) return;
  int j = t % cols, s = t / cols;
  Y[t] = f2bf(X[(size_t)s * cols + invp[j]]);
}

__global__ void permute_bf16(const bf16_t* __restrict__ X, const int* __restrict__ invp,
                             bf16_t* __restrict__ Y, int rows, int cols) {
  int t = blockIdx.x * blockDim.x + threadIdx.x;
  if (t >= rows * cols) return;
  int j = t % cols, s = t / cols;
  Y[t] = X[(size_t)s * cols + invp[j]];
}

// ---------- kernel 3: 2:4 sparse WMMA GEMM via V_SWMMAC ----------
// Y[s, o] = sum_j Wc(expand)[o, j] * X[s, j]     (X already permuted)
// A = compressed weight (sparse operand), B = X^T, 4 SWMMACs per K=64 step.
__global__ __launch_bounds__(256) void gemm_swmmac_bf16(
    const bf16_t*   __restrict__ Wc,   // O x K/2
    const unsigned* __restrict__ Idx,  // O x K/32
    const bf16_t*   __restrict__ X,    // S x K
    float*          __restrict__ Y,    // S x O
    int O, int S, int K) {
  const int lane = threadIdx.x & 31;
  const int wave = threadIdx.x >> 5;
  const int tilesPerRow = S >> 5;
  const int tile = blockIdx.x * 8 + wave;
  const int o0 = (tile / tilesPerRow) << 5;
  const int s0 = (tile % tilesPerRow) << 5;
  if (o0 >= O) return;
  const int lh = lane >> 4, ll = lane & 15;
  const int Kc = K >> 1, Kw = K >> 5;

  const bf16_t*   a0p = Wc + (size_t)(o0 + ll) * Kc + lh * 8;
  const bf16_t*   a1p = Wc + (size_t)(o0 + 16 + ll) * Kc + lh * 8;
  const unsigned* i0p = Idx + (size_t)(o0 + ll) * Kw + lh;
  const unsigned* i1p = Idx + (size_t)(o0 + 16 + ll) * Kw + lh;
  const bf16_t*   b0p = X + (size_t)(s0 + ll) * K + lh * 16;
  const bf16_t*   b1p = X + (size_t)(s0 + 16 + ll) * K + lh * 16;

  v8f c00 = {}, c01 = {}, c10 = {}, c11 = {};
  for (int kk = 0; kk < K; kk += 64) {
    __builtin_prefetch(a0p + (kk >> 1) + 128, 0, 0);   // global_prefetch_b8
    __builtin_prefetch(b0p + kk + 256, 0, 0);
    v16bf a0 = load_a(a0p + (kk >> 1));
    v16bf a1 = load_a(a1p + (kk >> 1));
    int ix0 = (int)i0p[kk >> 5];
    int ix1 = (int)i1p[kk >> 5];
    v32bf b0 = load_b64(b0p + kk);
    v32bf b1 = load_b64(b1p + kk);
    c00 = __builtin_amdgcn_swmmac_f32_16x16x64_bf16(false, a0, false, b0, c00, ix0, false, false);
    c01 = __builtin_amdgcn_swmmac_f32_16x16x64_bf16(false, a0, false, b1, c01, ix0, false, false);
    c10 = __builtin_amdgcn_swmmac_f32_16x16x64_bf16(false, a1, false, b0, c10, ix1, false, false);
    c11 = __builtin_amdgcn_swmmac_f32_16x16x64_bf16(false, a1, false, b1, c11, ix1, false, false);
  }
#pragma unroll
  for (int r = 0; r < 8; ++r) {
    int m = r + 8 * lh;                   // output-feature offset within tile
    Y[(size_t)(s0 + ll) * O + o0 + m]           = c00[r];
    Y[(size_t)(s0 + 16 + ll) * O + o0 + m]      = c01[r];
    Y[(size_t)(s0 + ll) * O + o0 + 16 + m]      = c10[r];
    Y[(size_t)(s0 + 16 + ll) * O + o0 + 16 + m] = c11[r];
  }
}

// ---------- kernel 4: RoPE (f32 in, bf16 out) ----------
__global__ void rope_bf16(const float* __restrict__ X,
                          const float* __restrict__ cosT,
                          const float* __restrict__ sinT,
                          bf16_t* __restrict__ Y, int nHeads) {
  int t = blockIdx.x * blockDim.x + threadIdx.x;
  int total = SEQ * nHeads * 64;
  if (t >= total) return;
  int d = t % 64, h = (t / 64) % nHeads, s = t / (64 * nHeads);
  int W = nHeads * HEAD_DIM;
  size_t base = (size_t)s * W + h * HEAD_DIM;
  float x1 = X[base + d], x2 = X[base + d + 64];
  float c1 = cosT[s * HEAD_DIM + d],      s1 = sinT[s * HEAD_DIM + d];
  float c2 = cosT[s * HEAD_DIM + d + 64], s2 = sinT[s * HEAD_DIM + d + 64];
  Y[base + d]      = f2bf(x1 * c1 - x2 * s1);
  Y[base + d + 64] = f2bf(x2 * c2 + x1 * s2);
}

// ---------- kernel 5: V -> bf16, d-major (Vt[(kvh*128+d)*SEQ + s]) ----------
__global__ void v_transpose(const float* __restrict__ V, bf16_t* __restrict__ Vt) {
  int t = blockIdx.x * blockDim.x + threadIdx.x;
  if (t >= SEQ * KVD) return;
  int col = t % KVD, s = t / KVD;
  Vt[(size_t)col * SEQ + s] = f2bf(V[t]);
}

// ---------- kernel 6: causal GQA flash attention, dense WMMA ----------
// grid = (SEQ/64, NUM_HEADS); 4 waves/block, wave = 16 q rows.
__global__ __launch_bounds__(128) void flash_attn(
    const bf16_t* __restrict__ Q,   // SEQ x 4096, rope'd
    const bf16_t* __restrict__ Kh,  // SEQ x 1024, rope'd
    const bf16_t* __restrict__ Vt,  // 1024 x SEQ, d-major
    bf16_t* __restrict__ O) {       // SEQ x 4096
  __shared__ bf16_t pbuf[4][16 * 32];
  const int lane = threadIdx.x & 31;
  const int wave = threadIdx.x >> 5;
  const int head = blockIdx.y;
  const int kvh  = head >> 2;
  const int q0   = blockIdx.x * 64 + wave * 16;
  const int lh = lane >> 4, ll = lane & 15;

  v16bf aq[4];
  {
    const bf16_t* qp = Q + (size_t)(q0 + ll) * HID + head * HEAD_DIM + lh * 8;
#pragma unroll
    for (int t = 0; t < 4; ++t) aq[t] = load_a(qp + t * 32);
  }

  v8f acc[8];
#pragma unroll
  for (int i = 0; i < 8; ++i) acc[i] = (v8f){};
  float rm[8], rl[8];
#pragma unroll
  for (int r = 0; r < 8; ++r) { rm[r] = -1e30f; rl[r] = 0.0f; }

  const float scale = 0.08838834764831845f;   // 1/sqrt(128)
  const int kend = q0 + 16;

  for (int kb = 0; kb < kend; kb += 32) {
    v8f sc[2]; sc[0] = (v8f){}; sc[1] = (v8f){};
#pragma unroll
    for (int t2 = 0; t2 < 2; ++t2) {
      const bf16_t* kp = Kh + (size_t)(kb + t2 * 16 + ll) * KVD + kvh * HEAD_DIM + lh * 8;
#pragma unroll
      for (int t = 0; t < 4; ++t) {
        v16bf bk = load_b(kp + t * 32);
        sc[t2] = __builtin_amdgcn_wmma_f32_16x16x32_bf16(false, aq[t], false, bk, (short)0, sc[t2], false, false);
      }
    }
    float sv[2][8], nm[8];
#pragma unroll
    for (int r = 0; r < 8; ++r) {
      int qrow = q0 + r + 8 * lh;
#pragma unroll
      for (int t2 = 0; t2 < 2; ++t2) {
        int key = kb + t2 * 16 + ll;
        float v = sc[t2][r] * scale;
        sv[t2][r] = (key <= qrow) ? v : -1e30f;
      }
      nm[r] = fmaxf(rm[r], fmaxf(sv[0][r], sv[1][r]));
    }
#pragma unroll
    for (int off = 1; off < 16; off <<= 1)
#pragma unroll
      for (int r = 0; r < 8; ++r)
        nm[r] = fmaxf(nm[r], __shfl_xor(nm[r], off, 32));
    float es[8], rs[8];
#pragma unroll
    for (int r = 0; r < 8; ++r) {
      es[r] = __expf(rm[r] - nm[r]);
      rm[r] = nm[r];
      float p0 = __expf(sv[0][r] - nm[r]);
      float p1 = __expf(sv[1][r] - nm[r]);
      sv[0][r] = p0; sv[1][r] = p1;
      rs[r] = p0 + p1;
    }
#pragma unroll
    for (int off = 1; off < 16; off <<= 1)
#pragma unroll
      for (int r = 0; r < 8; ++r)
        rs[r] += __shfl_xor(rs[r], off, 32);
#pragma unroll
    for (int r = 0; r < 8; ++r) rl[r] = rl[r] * es[r] + rs[r];
#pragma unroll
    for (int nt = 0; nt < 8; ++nt)
#pragma unroll
      for (int r = 0; r < 8; ++r) acc[nt][r] *= es[r];

    bf16_t* pb = pbuf[wave];
#pragma unroll
    for (int t2 = 0; t2 < 2; ++t2)
#pragma unroll
      for (int r = 0; r < 8; ++r)
        pb[(r + 8 * lh) * 32 + t2 * 16 + ll] = f2bf(sv[t2][r]);
    asm volatile("s_wait_dscnt 0" ::: "memory");
    v16bf pa = cat8(*(const v8bf*)(pb + ll * 32 + lh * 8),
                    *(const v8bf*)(pb + ll * 32 + 16 + lh * 8));

#pragma unroll
    for (int nt = 0; nt < 8; ++nt) {
      const bf16_t* vp = Vt + (size_t)(kvh * HEAD_DIM + nt * 16 + ll) * SEQ + kb + lh * 8;
      v16bf bv = load_b(vp);
      acc[nt] = __builtin_amdgcn_wmma_f32_16x16x32_bf16(false, pa, false, bv, (short)0, acc[nt], false, false);
    }
  }

#pragma unroll
  for (int nt = 0; nt < 8; ++nt)
#pragma unroll
    for (int r = 0; r < 8; ++r) {
      int m = r + 8 * lh;
      O[(size_t)(q0 + m) * HID + head * HEAD_DIM + nt * 16 + ll] = f2bf(acc[nt][r] / rl[r]);
    }
}

// ---------- host orchestration ----------
extern "C" void kernel_launch(void* const* d_in, const int* in_sizes, int n_in,
                              void* d_out, int out_size, void* d_ws, size_t ws_size,
                              hipStream_t stream) {
  (void)in_sizes; (void)n_in; (void)out_size; (void)ws_size;
  const float* hs   = (const float*)d_in[0];
  const float* wq   = (const float*)d_in[1];
  const float* wk   = (const float*)d_in[2];
  const float* wv   = (const float*)d_in[3];
  const float* wo   = (const float*)d_in[4];
  const float* sq   = (const float*)d_in[5];
  const float* sk   = (const float*)d_in[6];
  const float* svs  = (const float*)d_in[7];
  const float* so   = (const float*)d_in[8];
  const float* pqkv = (const float*)d_in[9];
  const float* po   = (const float*)d_in[10];
  const float* cosT = (const float*)d_in[11];
  const float* sinT = (const float*)d_in[12];
  float* out = (float*)d_out;

  char* ws = (char*)d_ws;
  auto carve = [&](size_t bytes) {
    char* p = ws;
    ws += (bytes + 255) & ~(size_t)255;
    return (void*)p;
  };
  int* inv_qkv = (int*)carve((size_t)HID * 4);
  int* inv_o   = (int*)carve((size_t)HID * 4);
  bf16_t*   hsP  = (bf16_t*)carve((size_t)SEQ * HID * 2);
  bf16_t*   wqc  = (bf16_t*)carve((size_t)HID * (HID / 2) * 2);
  bf16_t*   wkc  = (bf16_t*)carve((size_t)KVD * (HID / 2) * 2);
  bf16_t*   wvc  = (bf16_t*)carve((size_t)KVD * (HID / 2) * 2);
  bf16_t*   woc  = (bf16_t*)carve((size_t)HID * (HID / 2) * 2);
  unsigned* iq   = (unsigned*)carve((size_t)HID * (HID / 32) * 4);
  unsigned* ik   = (unsigned*)carve((size_t)KVD * (HID / 32) * 4);
  unsigned* iv   = (unsigned*)carve((size_t)KVD * (HID / 32) * 4);
  unsigned* io   = (unsigned*)carve((size_t)HID * (HID / 32) * 4);
  float*  qf  = (float*)carve((size_t)SEQ * HID * 4);
  float*  kf  = (float*)carve((size_t)SEQ * KVD * 4);
  float*  vf  = (float*)carve((size_t)SEQ * KVD * 4);
  bf16_t* qh  = (bf16_t*)carve((size_t)SEQ * HID * 2);
  bf16_t* kh  = (bf16_t*)carve((size_t)SEQ * KVD * 2);
  bf16_t* vt  = (bf16_t*)carve((size_t)KVD * SEQ * 2);
  bf16_t* aob = (bf16_t*)carve((size_t)SEQ * HID * 2);
  bf16_t* aobP = (bf16_t*)carve((size_t)SEQ * HID * 2);

  invert_perm<<<HID / 256, 256, 0, stream>>>(pqkv, inv_qkv, HID);
  invert_perm<<<HID / 256, 256, 0, stream>>>(po,   inv_o,   HID);

  int wQ = HID * (HID / 32), wKV = KVD * (HID / 32);
  compress_weights<<<wQ / 256,  256, 0, stream>>>(wq, sq,  inv_qkv, wqc, iq, HID, HID);
  compress_weights<<<wKV / 256, 256, 0, stream>>>(wk, sk,  inv_qkv, wkc, ik, KVD, HID);
  compress_weights<<<wKV / 256, 256, 0, stream>>>(wv, svs, inv_qkv, wvc, iv, KVD, HID);
  compress_weights<<<wQ / 256,  256, 0, stream>>>(wo, so,  inv_o,   woc, io, HID, HID);

  permute_cvt_bf16<<<(SEQ * HID) / 256, 256, 0, stream>>>(hs, inv_qkv, hsP, SEQ, HID);

  // sparse projections: tiles = (O/32)*(SEQ/32), 8 waves/block
  gemm_swmmac_bf16<<<(128 * 64) / 8, 256, 0, stream>>>(wqc, iq, hsP, qf, HID, SEQ, HID);
  gemm_swmmac_bf16<<<(32 * 64)  / 8, 256, 0, stream>>>(wkc, ik, hsP, kf, KVD, SEQ, HID);
  gemm_swmmac_bf16<<<(32 * 64)  / 8, 256, 0, stream>>>(wvc, iv, hsP, vf, KVD, SEQ, HID);

  rope_bf16<<<(SEQ * NUM_HEADS * 64) / 256, 256, 0, stream>>>(qf, cosT, sinT, qh, NUM_HEADS);
  rope_bf16<<<(SEQ * NUM_KV_HEADS * 64) / 256, 256, 0, stream>>>(kf, cosT, sinT, kh, NUM_KV_HEADS);
  v_transpose<<<(SEQ * KVD) / 256, 256, 0, stream>>>(vf, vt);

  dim3 ga(SEQ / 64, NUM_HEADS);
  flash_attn<<<ga, 128, 0, stream>>>(qh, kh, vt, aob);

  permute_bf16<<<(SEQ * HID) / 256, 256, 0, stream>>>(aob, inv_o, aobP, SEQ, HID);
  gemm_swmmac_bf16<<<(128 * 64) / 8, 256, 0, stream>>>(woc, io, aobP, out, HID, SEQ, HID);
}
